// MessagePassing_56547539419271
// MI455X (gfx1250) — matrix operand order repaired and checked by reading, not compiled, not run on hardware
//
#include <hip/hip_runtime.h>

typedef __bf16 bf16;
typedef __attribute__((ext_vector_type(16))) __bf16 v16bf;
typedef __attribute__((ext_vector_type(8)))  float  v8f;

#define Hdim 128
#define NT 8                    // n-tiles along the 128 output features
#define WAVE_REGION 13312       // per-wave LDS bytes (multiple of 256)
#define NWAVES 4                // waves per block
#define RS2 136                 // layer-2 A staging row stride (bf16 elems, padded)
#define XRS 132                 // LN staging row stride (f32 elems, padded)
#define XB_OFF 4608             // f32 LN buffer offset inside wave region
#define STATS_OFF 13056         // per-row mean/rstd offset inside wave region

// ---------- fragment helpers ----------

// A fragment (16x32 bf16, ISA layout): lane m=lane&15 holds row m;
// group g=lane>>4 selects K = k0+8g+{0..7} and k0+16+8g+{0..7}.
__device__ __forceinline__ v16bf load_afrag(const bf16* rowbase, int k0, int g) {
  union { v16bf v; uint4 q[2]; } u;
  u.q[0] = *(const uint4*)(rowbase + k0 + 8 * g);
  u.q[1] = *(const uint4*)(rowbase + k0 + 16 + 8 * g);
  return u.v;
}

// B fragment from pre-swizzled weights: fragment (c,t) is 32 lanes x 16 bf16 contiguous.
__device__ __forceinline__ v16bf load_bfrag(const bf16* wf, int c, int t, int lane) {
  union { v16bf v; uint4 q[2]; } u;
  const uint4* p = (const uint4*)(wf + (((c * NT + t) * 32 + lane) << 4));
  u.q[0] = p[0];
  u.q[1] = p[1];
  return u.v;
}

__device__ __forceinline__ void store4bf(bf16* p, float4 f) {
  union { bf16 h[4]; uint2 u; } t;
  t.h[0] = (bf16)f.x; t.h[1] = (bf16)f.y; t.h[2] = (bf16)f.z; t.h[3] = (bf16)f.w;
  *(uint2*)p = t.u;
}

// ---------- shared MLP + LayerNorm tile body ----------
// A1 (16 x K1 bf16, row stride RS1) already staged in wbase[0..).
template <int KC1>
__device__ __forceinline__ void mlp_ln_tile(
    char* wbase, int lane,
    const bf16* __restrict__ W1f, const bf16* __restrict__ W2f,
    const float* __restrict__ b1, const float* __restrict__ b2,
    const float* __restrict__ gamma, const float* __restrict__ beta,
    const float* __restrict__ resid_base, float* __restrict__ out_base, int RS1) {
  bf16*  A1    = (bf16*)wbase;
  bf16*  A2    = (bf16*)wbase;                 // reused after layer 1
  float* XB    = (float*)(wbase + XB_OFF);     // 16 x XRS f32
  float* stats = (float*)(wbase + STATS_OFF);  // [0..15]=mean, [16..31]=rstd
  const int m = lane & 15;
  const int g = lane >> 4;

  // ---- layer 1: (16 x K1) @ (K1 x 128) ----
  v8f acc[NT] = {};
  const bf16* arow = A1 + m * RS1;
  for (int c = 0; c < KC1; ++c) {
    v16bf a = load_afrag(arow, 32 * c, g);
#pragma unroll
    for (int t = 0; t < NT; ++t) {
      v16bf b = load_bfrag(W1f, c, t, lane);
      acc[t] = __builtin_amdgcn_wmma_f32_16x16x32_bf16(
          false, a, false, b, (short)0, acc[t], false, false);
    }
  }

  // ---- bias + relu -> A2 (16 x 128 bf16, stride RS2) ----
  // C layout: lane holds column N = 16t + m, rows M = v + 8g.
#pragma unroll
  for (int t = 0; t < NT; ++t) {
    float bias = b1[16 * t + m];
#pragma unroll
    for (int v = 0; v < 8; ++v) {
      float x = acc[t][v] + bias;
      x = x > 0.0f ? x : 0.0f;
      A2[(v + 8 * g) * RS2 + 16 * t + m] = (bf16)x;
    }
  }

  // ---- layer 2: (16 x 128) @ (128 x 128) ----
  v8f acc2[NT] = {};
  const bf16* arow2 = A2 + m * RS2;
  for (int c = 0; c < 4; ++c) {
    v16bf a = load_afrag(arow2, 32 * c, g);
#pragma unroll
    for (int t = 0; t < NT; ++t) {
      v16bf b = load_bfrag(W2f, c, t, lane);
      acc2[t] = __builtin_amdgcn_wmma_f32_16x16x32_bf16(
          false, a, false, b, (short)0, acc2[t], false, false);
    }
  }

  // ---- + bias -> fp32 LN staging ----
#pragma unroll
  for (int t = 0; t < NT; ++t) {
    float bias = b2[16 * t + m];
#pragma unroll
    for (int v = 0; v < 8; ++v)
      XB[(v + 8 * g) * XRS + 16 * t + m] = acc2[t][v] + bias;
  }

  // ---- per-row mean / rstd (lanes 0..15, one row each) ----
  if (lane < 16) {
    const float* row = XB + lane * XRS;
    float s = 0.0f, s2 = 0.0f;
    for (int j = 0; j < Hdim; j += 4) {
      float4 x = *(const float4*)(row + j);
      s  += x.x + x.y + x.z + x.w;
      s2 += x.x * x.x + x.y * x.y + x.z * x.z + x.w * x.w;
    }
    float mean = s * (1.0f / Hdim);
    float var  = s2 * (1.0f / Hdim) - mean * mean;
    stats[lane]      = mean;
    stats[16 + lane] = rsqrtf(var + 1e-5f);
  }

  // ---- normalize, affine, residual, coalesced store ----
  const int cc = lane * 4;
  float4 g4 = *(const float4*)(gamma + cc);
  float4 b4 = *(const float4*)(beta + cc);
  for (int r = 0; r < 16; ++r) {
    float mean = stats[r], rstd = stats[16 + r];
    float4 x4 = *(const float4*)(XB + r * XRS + cc);
    float4 f4 = *(const float4*)(resid_base + (size_t)r * Hdim + cc);
    float4 o;
    o.x = f4.x + (x4.x - mean) * rstd * g4.x + b4.x;
    o.y = f4.y + (x4.y - mean) * rstd * g4.y + b4.y;
    o.z = f4.z + (x4.z - mean) * rstd * g4.z + b4.z;
    o.w = f4.w + (x4.w - mean) * rstd * g4.w + b4.w;
    *(float4*)(out_base + (size_t)r * Hdim + cc) = o;
  }
}

// ---------- edge MLP ----------
__global__ __launch_bounds__(NWAVES * 32) void edge_mlp_kernel(
    const int* __restrict__ edge_idx, const float* __restrict__ node_feats,
    const float* __restrict__ edge_feats,
    const bf16* __restrict__ We1f, const bf16* __restrict__ We2f,
    const float* __restrict__ be1, const float* __restrict__ be2,
    const float* __restrict__ ge, const float* __restrict__ bge,
    float* __restrict__ out_edges, int ntiles) {
  __shared__ char smem[NWAVES * WAVE_REGION];
  const int wave = threadIdx.x >> 5;
  const int lane = threadIdx.x & 31;
  const int tile = blockIdx.x * NWAVES + wave;
  if (tile >= ntiles) return;                 // wave-uniform: EXEC stays all-ones
  char* wbase = smem + wave * WAVE_REGION;
  bf16* A1 = (bf16*)wbase;
  const int RS1 = 392;                        // 384 + pad (16B-aligned, conflict-free)
  const int e0 = tile * 16;
  const int cc = lane * 4;

  // stage A1 = [node[s] | node[r] | edge] as bf16, coalesced float4 gathers
  for (int r = 0; r < 16; ++r) {
    int e = e0 + r;
    int s = edge_idx[2 * e];
    int d = edge_idx[2 * e + 1];
    float4 fs = *(const float4*)(node_feats + (size_t)s * Hdim + cc);
    float4 fr = *(const float4*)(node_feats + (size_t)d * Hdim + cc);
    float4 fe = *(const float4*)(edge_feats + (size_t)e * Hdim + cc);
    bf16* row = A1 + r * RS1;
    store4bf(row + cc, fs);
    store4bf(row + 128 + cc, fr);
    store4bf(row + 256 + cc, fe);
  }
  mlp_ln_tile<12>(wbase, lane, We1f, We2f, be1, be2, ge, bge,
                  edge_feats + (size_t)e0 * Hdim,
                  out_edges + (size_t)e0 * Hdim, RS1);
}

// ---------- node MLP ----------
__global__ __launch_bounds__(NWAVES * 32) void node_mlp_kernel(
    const float* __restrict__ node_feats, const float* __restrict__ acc,
    const bf16* __restrict__ Wn1f, const bf16* __restrict__ Wn2f,
    const float* __restrict__ bn1, const float* __restrict__ bn2,
    const float* __restrict__ gn, const float* __restrict__ bgn,
    float* __restrict__ out_nodes, int ntiles) {
  __shared__ char smem[NWAVES * WAVE_REGION];
  const int wave = threadIdx.x >> 5;
  const int lane = threadIdx.x & 31;
  const int tile = blockIdx.x * NWAVES + wave;
  if (tile >= ntiles) return;
  char* wbase = smem + wave * WAVE_REGION;
  bf16* A1 = (bf16*)wbase;
  const int RS1 = 264;                        // 256 + pad
  const int n0 = tile * 16;
  const int cc = lane * 4;

  for (int r = 0; r < 16; ++r) {
    int n = n0 + r;
    float4 fn = *(const float4*)(node_feats + (size_t)n * Hdim + cc);
    float4 fa = *(const float4*)(acc + (size_t)n * Hdim + cc);
    bf16* row = A1 + r * RS1;
    store4bf(row + cc, fn);
    store4bf(row + 128 + cc, fa);
  }
  mlp_ln_tile<8>(wbase, lane, Wn1f, Wn2f, bn1, bn2, gn, bgn,
                 node_feats + (size_t)n0 * Hdim,
                 out_nodes + (size_t)n0 * Hdim, RS1);
}

// ---------- segment sum: acc[r[e]] += edge_feats[e] ----------
__global__ void segsum_kernel(const int* __restrict__ edge_idx,
                              const float* __restrict__ edge_feats,
                              float* __restrict__ acc, int E) {
  int id = blockIdx.x * blockDim.x + threadIdx.x;
  if (id >= E * 32) return;
  int e  = id >> 5;
  int c4 = (id & 31) * 4;
  int r  = edge_idx[2 * e + 1];
  float4 v = *(const float4*)(edge_feats + (size_t)e * Hdim + c4);
  float* dst = acc + (size_t)r * Hdim + c4;
  unsafeAtomicAdd(dst + 0, v.x);
  unsafeAtomicAdd(dst + 1, v.y);
  unsafeAtomicAdd(dst + 2, v.z);
  unsafeAtomicAdd(dst + 3, v.w);
}

// ---------- weight fp32 -> bf16, swizzled into B-fragment order ----------
__global__ void wconv_kernel(const float* __restrict__ src, bf16* __restrict__ dst, int K) {
  int id = blockIdx.x * blockDim.x + threadIdx.x;
  if (id >= K * Hdim) return;
  int j    = id & 15;          // value within lane: j = 2v + p
  int lane = (id >> 4) & 31;
  int t    = (id >> 9) & 7;    // n-tile
  int c    = id >> 12;         // k-chunk
  int v = j >> 1, p = j & 1;
  int k = 32 * c + ((lane & 16) >> 1) + (j & 7) + ((j >> 3) << 4);   // lanes 16..31 carry K+8 (and K+24)
  int n = 16 * t + (lane & 15);
  dst[id] = (bf16)src[k * Hdim + n];
}

// ---------- host launcher ----------
extern "C" void kernel_launch(void* const* d_in, const int* in_sizes, int n_in,
                              void* d_out, int out_size, void* d_ws, size_t ws_size,
                              hipStream_t stream) {
  const int Nn = 50000, Ee = 500000;
  const int*   edge_idx   = (const int*)d_in[0];
  const float* node_feats = (const float*)d_in[1];
  const float* edge_feats = (const float*)d_in[2];
  const float* We1 = (const float*)d_in[3];
  const float* be1 = (const float*)d_in[4];
  const float* We2 = (const float*)d_in[5];
  const float* be2 = (const float*)d_in[6];
  const float* ge  = (const float*)d_in[7];
  const float* bge = (const float*)d_in[8];
  const float* Wn1 = (const float*)d_in[9];
  const float* bn1 = (const float*)d_in[10];
  const float* Wn2 = (const float*)d_in[11];
  const float* bn2 = (const float*)d_in[12];
  const float* gn  = (const float*)d_in[13];
  const float* bgn = (const float*)d_in[14];

  // workspace: fp32 accumulator + 4 swizzled bf16 weight blobs
  float* acc = (float*)d_ws;
  size_t off = (size_t)Nn * Hdim * sizeof(float);
  bf16* We1f = (bf16*)((char*)d_ws + off); off += 384 * 128 * sizeof(bf16);
  bf16* We2f = (bf16*)((char*)d_ws + off); off += 128 * 128 * sizeof(bf16);
  bf16* Wn1f = (bf16*)((char*)d_ws + off); off += 256 * 128 * sizeof(bf16);
  bf16* Wn2f = (bf16*)((char*)d_ws + off); off += 128 * 128 * sizeof(bf16);

  hipMemsetAsync(acc, 0, (size_t)Nn * Hdim * sizeof(float), stream);
  wconv_kernel<<<(384 * 128 + 255) / 256, 256, 0, stream>>>(We1, We1f, 384);
  wconv_kernel<<<(128 * 128 + 255) / 256, 256, 0, stream>>>(We2, We2f, 128);
  wconv_kernel<<<(256 * 128 + 255) / 256, 256, 0, stream>>>(Wn1, Wn1f, 256);
  wconv_kernel<<<(128 * 128 + 255) / 256, 256, 0, stream>>>(Wn2, Wn2f, 128);
  segsum_kernel<<<(Ee * 32 + 255) / 256, 256, 0, stream>>>(edge_idx, edge_feats, acc, Ee);

  float* out_nodes = (float*)d_out;
  float* out_edges = out_nodes + (size_t)Nn * Hdim;
  const int etiles = Ee / 16;   // 31250
  const int ntiles = Nn / 16;   // 3125

  edge_mlp_kernel<<<(etiles + NWAVES - 1) / NWAVES, NWAVES * 32, 0, stream>>>(
      edge_idx, node_feats, edge_feats, We1f, We2f, be1, be2, ge, bge, out_edges, etiles);
  node_mlp_kernel<<<(ntiles + NWAVES - 1) / NWAVES, NWAVES * 32, 0, stream>>>(
      node_feats, acc, Wn1f, Wn2f, bn1, bn2, gn, bgn, out_nodes, ntiles);
}
